// GraphAllEdgeNet_85495618994963
// MI455X (gfx1250) — matrix operand order repaired and checked by reading, not compiled, not run on hardware
//
#include <hip/hip_runtime.h>

// ---------------- problem constants ----------------
#define N_NODES   100000
#define N_EDGES   800000
#define CH        64
#define BN_SCALE  0.9999950000374997f   // 1/sqrt(1+1e-5)

typedef __bf16 bf16_t;
typedef __attribute__((ext_vector_type(16))) __bf16 v16bf;
typedef __attribute__((ext_vector_type(8)))  float  v8f;

struct B32 { uint4 lo, hi; };   // 32 bytes == v16bf

static __device__ __forceinline__ bf16_t f2bf(float f) {
  unsigned u = __builtin_bit_cast(unsigned, f);
  u += 0x7FFFu + ((u >> 16) & 1u);              // round-to-nearest-even
  unsigned short h = (unsigned short)(u >> 16);
  return __builtin_bit_cast(bf16_t, h);
}

// Load a 16-half fragment as two 16B chunks (2x ds_load_b128).
static __device__ __forceinline__ v16bf load_frag16(const bf16_t* p0, const bf16_t* p1) {
  B32 t;
  t.lo = *(const uint4*)p0;
  t.hi = *(const uint4*)p1;
  return __builtin_bit_cast(v16bf, t);
}

// A-fragment for v_wmma_f32_16x16x32_bf16 from a row-major tile.
// row = &tile[M][kbase] (16B aligned). Lane half pattern: K = hi*16 + {0..15}.
static __device__ __forceinline__ v16bf load_a_frag(const bf16_t* row, int hi) {
  return load_frag16(row + hi * 16, row + hi * 16 + 8);
}

// B-fragment from weight storage pre-swizzled to fragment order:
// element (K,N): frag = (K/32)*4 + N/16 ; lane = ((K/16)&1)*16 + (N&15) ; h = K&15
// dst = frag*512 + lane*16 + h  -> each lane's 16 halfs are 32 contiguous bytes.
static __device__ __forceinline__ v16bf load_b_frag_sw(const bf16_t* base, int frag, int lane) {
  const bf16_t* p = base + ((size_t)frag << 9) + (lane << 4);
  return load_frag16(p, p + 8);
}

static __device__ __forceinline__ v8f v8zero() {
  v8f z = {0.f, 0.f, 0.f, 0.f, 0.f, 0.f, 0.f, 0.f};
  return z;
}

// CDNA5 async global->LDS DMA, 16 bytes per lane, tracked by ASYNCcnt.
// GV mode: 64-bit global address in VGPR pair, LDS byte offset in VDST VGPR.
// (The LDS offset is the low 32 bits of the flat pointer: LDS aperture maps
//  addr[31:0] directly to the LDS address.)
static __device__ __forceinline__ void async_copy_b128(unsigned lds_off, const void* gaddr) {
  asm volatile("global_load_async_to_lds_b128 %0, %1, off"
               :: "v"(lds_off), "v"(gaddr) : "memory");
}
static __device__ __forceinline__ void wait_async0() {
  asm volatile("s_wait_asynccnt 0x0" ::: "memory");
}
static __device__ __forceinline__ unsigned lds_off32(const void* p) {
  return (unsigned)(size_t)p;
}

// ---------------- weight conversion f32 -> bf16, swizzled to B-fragment order ----
// Handles nmat stacked [Kdim][64] matrices; each swizzled independently.
__global__ void cvt_swz_kernel(const float* __restrict__ in,
                               unsigned short* __restrict__ out, int Kdim, int total) {
  const int per = Kdim * 64;
  for (int i = blockIdx.x * blockDim.x + threadIdx.x; i < total;
       i += gridDim.x * blockDim.x) {
    int mat = i / per, rem = i - mat * per;
    int K = rem >> 6, N = rem & 63;
    int ks = K >> 5, kh = (K >> 4) & 1, h = K & 15;
    int nt = N >> 4, n = N & 15;
    int dst = mat * per + (((ks << 2) + nt) << 9) + (((kh << 4) + n) << 4) + h;
    unsigned u = __builtin_bit_cast(unsigned, in[i]);
    u += 0x7FFFu + ((u >> 16) & 1u);
    out[dst] = (unsigned short)(u >> 16);
  }
}

// ---------------- input projection: gf = relu(bn0(x@W0 + b0)) ----------------
// phase 0: even nodes (audio, slab 0, W0a) ; phase 1: odd nodes (video, slab 1, W0v)
__global__ void proj_kernel(const float* __restrict__ x,           // [N,2,512]
                            const unsigned short* __restrict__ W0a_sw,
                            const unsigned short* __restrict__ W0v_sw,
                            const float* __restrict__ b0a, const float* __restrict__ b0v,
                            const float* __restrict__ bn0g, const float* __restrict__ bn0b,
                            float* __restrict__ gf, int ntiles) {
  __shared__ __align__(16) bf16_t sB[4 * 512];      // 4 B-fragments of current k-step
  __shared__ __align__(16) bf16_t sA[4][16 * 32];   // per-wave A tiles
  __shared__ float sbias[64], sg[64], sbb[64];

  const int tid   = threadIdx.x;        // 128 threads, 4 waves
  const int wave  = tid >> 5;
  const int lane  = tid & 31;
  const int phase = blockIdx.y;         // 0=audio, 1=video
  const unsigned short* W = phase ? W0v_sw : W0a_sw;
  const float* b0 = phase ? b0v : b0a;

  if (tid < 64) {
    sbias[tid] = b0[tid];
    sg[tid]    = BN_SCALE * bn0g[tid];
    sbb[tid]   = bn0b[tid];
  }

  const int  tile  = blockIdx.x * 4 + wave;
  const bool valid = (tile < ntiles);

  v8f accv[4];
#pragma unroll
  for (int nt = 0; nt < 4; ++nt) accv[nt] = v8zero();

  const int M = lane & 15, hi = lane >> 4;
  const unsigned sB_off = lds_off32(sB);

  for (int ks = 0; ks < 16; ++ks) {          // K = 512 = 16 * 32
    __syncthreads();
    // stage B fragments for this k-step via async DMA (2048 bf16, contiguous)
    {
      const char* src = (const char*)(W + (size_t)ks * 2048);
      for (int i = tid; i < 256; i += 128)
        async_copy_b128(sB_off + i * 16, src + (size_t)i * 16);
    }
    // stage per-wave A chunk [16 x 32] from x (f32 -> bf16), float4-vectorized
    for (int idx4 = lane; idx4 < 16 * 8; idx4 += 32) {
      int r = idx4 >> 3, c = (idx4 & 7) << 2;
      int t = valid ? (tile * 16 + r) : 0;
      int n = 2 * t + phase;
      float4 v = *(const float4*)(x + ((size_t)n * 2 + phase) * 512 + ks * 32 + c);
      union { bf16_t b[4]; uint2 u; } pk;
      pk.b[0] = f2bf(v.x); pk.b[1] = f2bf(v.y);
      pk.b[2] = f2bf(v.z); pk.b[3] = f2bf(v.w);
      *(uint2*)&sA[wave][r * 32 + c] = pk.u;
    }
    wait_async0();
    __syncthreads();

    v16bf A = load_a_frag(&sA[wave][M * 32], hi);
    v16bf Bf[4];
#pragma unroll
    for (int nt = 0; nt < 4; ++nt) Bf[nt] = load_b_frag_sw(sB, nt, lane);
#pragma unroll
    for (int nt = 0; nt < 4; ++nt)
      accv[nt] = __builtin_amdgcn_wmma_f32_16x16x32_bf16(
          false, A, false, Bf[nt], (short)0, accv[nt], false, false);
  }

  if (valid) {
    const int Nn = lane & 15;
#pragma unroll
    for (int nt = 0; nt < 4; ++nt) {
      int c = nt * 16 + Nn;
      float s = sg[c], bb = sbb[c], bias = sbias[c];
#pragma unroll
      for (int r = 0; r < 8; ++r) {
        int Mr = r + 8 * hi;
        int n = 2 * (tile * 16 + Mr) + phase;
        float v = accv[nt][r] + bias;
        v = v * s + bb;
        v = v > 0.f ? v : 0.f;
        gf[(size_t)n * CH + c] = v;
      }
    }
  }
}

// ---------------- EdgeConv message + scatter ----------------
// One wave handles a 16-edge tile:  msg = relu(bn1([xi | xj-xi])) [16x128]
//   GEMM1 (16x128 @ 128x64) -> bn2+relu -> GEMM2 (16x64 @ 64x64) -> mask -> atomic scatter
__global__ void edge_kernel(const float* __restrict__ Xin,          // [N,64]
                            const int* __restrict__ eidx,           // [2,E]
                            const int* __restrict__ edelta,
                            const int* __restrict__ eself,
                            const unsigned short* __restrict__ W1sw,// swizzled [128][64]
                            const unsigned short* __restrict__ W2sw,// swizzled [64][64]
                            const float* __restrict__ g1, const float* __restrict__ b1,
                            const float* __restrict__ g2, const float* __restrict__ b2,
                            float* __restrict__ acc, float* __restrict__ cnt,
                            int maskSel, int E) {
  __shared__ __align__(16) bf16_t sW1[128 * 64];    // 16 B-fragments
  __shared__ __align__(16) bf16_t sW2[64 * 64];     // 8 B-fragments
  __shared__ __align__(16) float  sg1[128], sb1[128];
  __shared__ float  sg2[64], sb2[64];
  __shared__ __align__(16) bf16_t sMsg[4][16 * 128];
  __shared__ __align__(16) bf16_t sH[4][16 * 64];
  __shared__ int    sDst[4][16], sSrc[4][16];
  __shared__ float  sMask[4][16];

  const int tid  = threadIdx.x;        // 128 threads = 4 waves
  const int wave = tid >> 5;
  const int lane = tid & 31;
  const int tile = blockIdx.x * 4 + wave;   // grid sized so tile < E/16 always

  // stage weights via async global->LDS DMA (already swizzled; contiguous copy)
  {
    const unsigned w1o = lds_off32(sW1), w2o = lds_off32(sW2);
    for (int i = tid; i < 1024; i += 128)
      async_copy_b128(w1o + i * 16, (const char*)W1sw + (size_t)i * 16);
    for (int i = tid; i < 512; i += 128)
      async_copy_b128(w2o + i * 16, (const char*)W2sw + (size_t)i * 16);
  }
  if (tid < 128) { sg1[tid] = BN_SCALE * g1[tid]; sb1[tid] = b1[tid]; }
  if (tid < 64)  { sg2[tid] = BN_SCALE * g2[tid]; sb2[tid] = b2[tid]; }

  // per-tile edge metadata + count scatter
  if (lane < 16) {
    int e = tile * 16 + lane;
    int s = eidx[e], d = eidx[E + e];
    sSrc[wave][lane] = s;
    sDst[wave][lane] = d;
    int dd = edelta[e], sf = eself[e];
    float m = (maskSel == 0)
                  ? (dd < 1 ? 1.f : 0.f)
                  : ((((dd >= 1) && (dd < 4)) || (sf == 1)) ? 1.f : 0.f);
    sMask[wave][lane] = m;
    if (m != 0.f) atomicAdd(&cnt[d], m);
  }
  wait_async0();
  __syncthreads();

  // build message tile [16 x 128] bf16 after BN1+ReLU (float4 gathers, b64 LDS stores)
  for (int idx4 = lane; idx4 < 16 * 32; idx4 += 32) {
    int r = idx4 >> 5, c = (idx4 & 31) << 2;    // c = 0..124
    int cc = c & 63;
    int ni = sDst[wave][r];
    float4 v = *(const float4*)(Xin + (size_t)ni * CH + cc);
    if (c >= 64) {
      int nj = sSrc[wave][r];
      float4 vj = *(const float4*)(Xin + (size_t)nj * CH + cc);
      v.x = vj.x - v.x; v.y = vj.y - v.y; v.z = vj.z - v.z; v.w = vj.w - v.w;
    }
    float4 s = *(const float4*)(sg1 + c);
    float4 bb = *(const float4*)(sb1 + c);
    float o0 = v.x * s.x + bb.x, o1 = v.y * s.y + bb.y;
    float o2 = v.z * s.z + bb.z, o3 = v.w * s.w + bb.w;
    union { bf16_t b[4]; uint2 u; } pk;
    pk.b[0] = f2bf(o0 > 0.f ? o0 : 0.f);
    pk.b[1] = f2bf(o1 > 0.f ? o1 : 0.f);
    pk.b[2] = f2bf(o2 > 0.f ? o2 : 0.f);
    pk.b[3] = f2bf(o3 > 0.f ? o3 : 0.f);
    *(uint2*)&sMsg[wave][r * 128 + c] = pk.u;
  }
  __syncthreads();

  const int M = lane & 15, hi = lane >> 4, Nn = lane & 15;

  // GEMM1: msg(16x128) @ W1(128x64)
  v8f h1[4];
#pragma unroll
  for (int nt = 0; nt < 4; ++nt) h1[nt] = v8zero();
#pragma unroll
  for (int ks = 0; ks < 4; ++ks) {
    v16bf A = load_a_frag(&sMsg[wave][M * 128 + ks * 32], hi);
    v16bf Bf[4];
#pragma unroll
    for (int nt = 0; nt < 4; ++nt) Bf[nt] = load_b_frag_sw(sW1, ks * 4 + nt, lane);
#pragma unroll
    for (int nt = 0; nt < 4; ++nt)
      h1[nt] = __builtin_amdgcn_wmma_f32_16x16x32_bf16(
          false, A, false, Bf[nt], (short)0, h1[nt], false, false);
  }

  // BN2 + ReLU, transpose to A-layout via LDS
#pragma unroll
  for (int nt = 0; nt < 4; ++nt) {
    int c = nt * 16 + Nn;
    float s = sg2[c], bb = sb2[c];
#pragma unroll
    for (int r = 0; r < 8; ++r) {
      float v = h1[nt][r] * s + bb;
      v = v > 0.f ? v : 0.f;
      sH[wave][(r + 8 * hi) * 64 + c] = f2bf(v);
    }
  }
  __syncthreads();

  // GEMM2: h(16x64) @ W2(64x64)
  v8f outv[4];
#pragma unroll
  for (int nt = 0; nt < 4; ++nt) outv[nt] = v8zero();
#pragma unroll
  for (int ks = 0; ks < 2; ++ks) {
    v16bf A = load_a_frag(&sH[wave][M * 64 + ks * 32], hi);
    v16bf Bf[4];
#pragma unroll
    for (int nt = 0; nt < 4; ++nt) Bf[nt] = load_b_frag_sw(sW2, ks * 4 + nt, lane);
#pragma unroll
    for (int nt = 0; nt < 4; ++nt)
      outv[nt] = __builtin_amdgcn_wmma_f32_16x16x32_bf16(
          false, A, false, Bf[nt], (short)0, outv[nt], false, false);
  }

  // masked scatter-add into accumulator
#pragma unroll
  for (int r = 0; r < 8; ++r) {
    int Mr = r + 8 * hi;
    float m = sMask[wave][Mr];
    if (m != 0.f) {
      int n = sDst[wave][Mr];
#pragma unroll
      for (int nt = 0; nt < 4; ++nt)
        atomicAdd(&acc[(size_t)n * CH + nt * 16 + Nn], outv[nt][r] * m);
    }
  }
}

// ---------------- masked-mean + residual + BN + ReLU ----------------
__global__ void finalize_kernel(const float* __restrict__ acc, const float* __restrict__ cnt,
                                const float* __restrict__ res,
                                const float* __restrict__ g, const float* __restrict__ b,
                                float* __restrict__ out, int useBN, int useRelu, int total) {
  int i = blockIdx.x * blockDim.x + threadIdx.x;
  if (i >= total) return;
  int n = i >> 6, c = i & 63;
  float cv = cnt[n];
  cv = cv > 1.f ? cv : 1.f;
  float v = acc[i] / cv;
  if (res) v += res[i];
  if (useBN) v = v * (BN_SCALE * g[c]) + b[c];
  if (useRelu) v = v > 0.f ? v : 0.f;
  out[i] = v;
}

// ---------------- heads ----------------
// d_out layout: out[N,2] | audio_out[N/2,2] | video_out[N/2,2]
__global__ void heads_kernel(const float* __restrict__ gf, const float* __restrict__ g4,
                             const float* __restrict__ fcW,  const float* __restrict__ fcb,
                             const float* __restrict__ fcaW, const float* __restrict__ fcab,
                             const float* __restrict__ fcvW, const float* __restrict__ fcvb,
                             float* __restrict__ out, int N) {
  int n = blockIdx.x * blockDim.x + threadIdx.x;
  if (n >= N) return;
  const float* r4 = g4 + (size_t)n * CH;
  float o0 = fcb[0], o1 = fcb[1];
#pragma unroll 8
  for (int c = 0; c < CH; ++c) {
    float v = r4[c];
    o0 += v * fcW[c * 2 + 0];
    o1 += v * fcW[c * 2 + 1];
  }
  out[(size_t)n * 2 + 0] = o0;
  out[(size_t)n * 2 + 1] = o1;

  const float* rg = gf + (size_t)n * CH;
  int row = n >> 1;
  if ((n & 1) == 0) {
    float a0 = fcab[0], a1 = fcab[1];
#pragma unroll 8
    for (int c = 0; c < CH; ++c) {
      float v = rg[c];
      a0 += v * fcaW[c * 2 + 0];
      a1 += v * fcaW[c * 2 + 1];
    }
    size_t base = (size_t)N * 2;
    out[base + row * 2 + 0] = a0;
    out[base + row * 2 + 1] = a1;
  } else {
    float a0 = fcvb[0], a1 = fcvb[1];
#pragma unroll 8
    for (int c = 0; c < CH; ++c) {
      float v = rg[c];
      a0 += v * fcvW[c * 2 + 0];
      a1 += v * fcvW[c * 2 + 1];
    }
    size_t base = (size_t)N * 2 + (size_t)(N / 2) * 2;
    out[base + row * 2 + 0] = a0;
    out[base + row * 2 + 1] = a1;
  }
}

// ---------------- host ----------------
extern "C" void kernel_launch(void* const* d_in, const int* in_sizes, int n_in,
                              void* d_out, int out_size, void* d_ws, size_t ws_size,
                              hipStream_t stream) {
  (void)in_sizes; (void)n_in; (void)out_size; (void)ws_size;

  const float* x          = (const float*)d_in[0];
  const int*   edge_index = (const int*)  d_in[1];
  const int*   edge_delta = (const int*)  d_in[2];
  const int*   edge_self  = (const int*)  d_in[3];
  // d_in[4] audio_node_mask == (n % 2 == 0), recomputed on device
  const float* W0a   = (const float*)d_in[5];
  const float* b0a   = (const float*)d_in[6];
  const float* W0v   = (const float*)d_in[7];
  const float* b0v   = (const float*)d_in[8];
  const float* bn0_g = (const float*)d_in[9];
  const float* bn0_b = (const float*)d_in[10];
  const float* ec_bn1_g = (const float*)d_in[11];
  const float* ec_bn1_b = (const float*)d_in[12];
  const float* ec_fc1_W = (const float*)d_in[13];
  const float* ec_bn2_g = (const float*)d_in[14];
  const float* ec_bn2_b = (const float*)d_in[15];
  const float* ec_fc2_W = (const float*)d_in[16];
  const float* bn_g  = (const float*)d_in[17];
  const float* bn_b  = (const float*)d_in[18];
  const float* fca_W = (const float*)d_in[19];
  const float* fca_b = (const float*)d_in[20];
  const float* fcv_W = (const float*)d_in[21];
  const float* fcv_b = (const float*)d_in[22];
  const float* fc_W  = (const float*)d_in[23];
  const float* fc_b  = (const float*)d_in[24];

  const int N = N_NODES, E = N_EDGES;

  // workspace carve-up (256B aligned)
  char* ws = (char*)d_ws;
  size_t off = 0;
  auto take = [&](size_t bytes) -> void* {
    void* p = ws + off;
    off = (off + bytes + 255) & ~(size_t)255;
    return p;
  };
  float* gf  = (float*)take((size_t)N * CH * 4);
  float* Xb  = (float*)take((size_t)N * CH * 4);   // block input / residual / output
  float* Hb  = (float*)take((size_t)N * CH * 4);   // intra-block intermediate
  float* ACC = (float*)take((size_t)N * CH * 4);
  float* CNT = (float*)take((size_t)N * 4);
  unsigned short* W0a_sw = (unsigned short*)take((size_t)512 * 64 * 2);
  unsigned short* W0v_sw = (unsigned short*)take((size_t)512 * 64 * 2);
  unsigned short* FC1_sw = (unsigned short*)take((size_t)4 * 128 * 64 * 2);
  unsigned short* FC2_sw = (unsigned short*)take((size_t)4 * 64 * 64 * 2);

  // weight conversion + swizzle to B-fragment order
  cvt_swz_kernel<<<128, 256, 0, stream>>>(W0a, W0a_sw, 512, 512 * 64);
  cvt_swz_kernel<<<128, 256, 0, stream>>>(W0v, W0v_sw, 512, 512 * 64);
  cvt_swz_kernel<<<128, 256, 0, stream>>>(ec_fc1_W, FC1_sw, 128, 4 * 128 * 64);
  cvt_swz_kernel<<<64, 256, 0, stream>>>(ec_fc2_W, FC2_sw, 64, 4 * 64 * 64);

  // input projection + BN0 + ReLU
  const int ntiles = (N / 2) / 16;                       // 3125 per parity
  proj_kernel<<<dim3((ntiles + 3) / 4, 2), 128, 0, stream>>>(
      x, W0a_sw, W0v_sw, b0a, b0v, bn0_g, bn0_b, gf, ntiles);

  const int edgeBlocks = E / 64;                         // 4 tiles of 16 edges per block
  const int finBlocks  = (N * CH + 255) / 256;

  for (int k = 0; k < 4; ++k) {
    const unsigned short* W1 = FC1_sw + (size_t)k * 128 * 64;
    const unsigned short* W2 = FC2_sw + (size_t)k * 64 * 64;
    const float* g1 = ec_bn1_g + k * 128;
    const float* b1 = ec_bn1_b + k * 128;
    const float* g2 = ec_bn2_g + k * 64;
    const float* b2 = ec_bn2_b + k * 64;
    const float* xin = (k == 0) ? gf : Xb;

    // pass 1 (mask m1) -> Hb (plain masked mean)
    hipMemsetAsync(ACC, 0, (size_t)N * CH * 4, stream);
    hipMemsetAsync(CNT, 0, (size_t)N * 4, stream);
    edge_kernel<<<edgeBlocks, 128, 0, stream>>>(
        xin, edge_index, edge_delta, edge_self, W1, W2, g1, b1, g2, b2,
        ACC, CNT, 0, E);
    finalize_kernel<<<finBlocks, 256, 0, stream>>>(
        ACC, CNT, nullptr, nullptr, nullptr, Hb, 0, 0, N * CH);

    // pass 2 (mask m2) -> Xb with block epilogue
    hipMemsetAsync(ACC, 0, (size_t)N * CH * 4, stream);
    hipMemsetAsync(CNT, 0, (size_t)N * 4, stream);
    edge_kernel<<<edgeBlocks, 128, 0, stream>>>(
        Hb, edge_index, edge_delta, edge_self, W1, W2, g1, b1, g2, b2,
        ACC, CNT, 1, E);
    const float* res = (k == 0) ? nullptr : Xb;
    const int useBN = (k < 3), useRelu = (k < 3);
    const float* g = useBN ? (bn_g + k * 64) : nullptr;
    const float* b = useBN ? (bn_b + k * 64) : nullptr;
    finalize_kernel<<<finBlocks, 256, 0, stream>>>(
        ACC, CNT, res, g, b, Xb, useBN, useRelu, N * CH);
  }

  // heads
  heads_kernel<<<(N + 255) / 256, 256, 0, stream>>>(
      gf, Xb, fc_W, fc_b, fca_W, fca_b, fcv_W, fcv_b, (float*)d_out, N);
}
